// SeparableFC_20650202759866
// MI455X (gfx1250) — compile-verified, matrix operand order checked by reading, not run on hardware
//
#include <hip/hip_runtime.h>
#include <hip/hip_bf16.h>

typedef __bf16 bf16_t;
typedef __attribute__((ext_vector_type(16))) __bf16 v16bf;
typedef __attribute__((ext_vector_type(8)))  float  v8f;

#define B_ 1024
#define L_ 2048
#define C_ 32
#define O_ 2048

// ---------------------------------------------------------------------------
// Kernel 1: x (B,L,C) f32  ->  xT (B,C,L) bf16   (LDS tile transpose)
// ---------------------------------------------------------------------------
__global__ void __launch_bounds__(256) convert_x_kernel(
    const float* __restrict__ x, bf16_t* __restrict__ xT) {
  __shared__ bf16_t tile[32 * 66];  // pad 64->66 halfs: conflict-free banks
  const int b   = blockIdx.y;
  const int l0  = blockIdx.x * 64;
  const int tid = threadIdx.x;

  const float* xb = x + (size_t)b * (L_ * C_) + (size_t)l0 * C_;
  // coalesced f32 reads: i walks (l,c) contiguously
  for (int i = tid; i < 64 * 32; i += 256) {
    const int l = i >> 5, c = i & 31;
    tile[c * 66 + l] = (bf16_t)xb[i];
  }
  __syncthreads();

  bf16_t* xtb = xT + (size_t)b * (C_ * L_);
  // coalesced bf16 writes: i walks (c,l) contiguously
  for (int i = tid; i < 32 * 64; i += 256) {
    const int c = i >> 6, l = i & 63;
    xtb[(size_t)c * L_ + l0 + l] = tile[c * 66 + l];
  }
}

// ---------------------------------------------------------------------------
// Kernel 2: W_pos (O,L) f32 -> bf16
// ---------------------------------------------------------------------------
__global__ void __launch_bounds__(256) convert_w_kernel(
    const float* __restrict__ w, bf16_t* __restrict__ wb, int n) {
  const int i = blockIdx.x * 256 + threadIdx.x;
  if (i < n) wb[i] = (bf16_t)w[i];
}

// ---------------------------------------------------------------------------
// Kernel 3: fused batched GEMM + channel contraction, 4x N-blocked.
// Per wave: (b, 64-wide o-block) = 4 o-tiles x 2 c-tiles = 8 accumulators.
//   u[c,o] = sum_l xT[b,c,l] * W_pos[o,l]      (v_wmma_f32_16x16x32_bf16)
//   out[b,o] = sum_c u[c,o] * W_chan[o,c]      (VALU epilogue + shfl_xor)
// A operands (x) are shared across the 4 o-tiles -> 1.5 VMEM per WMMA.
// ---------------------------------------------------------------------------
__global__ void __launch_bounds__(256) sepfc_main_kernel(
    const bf16_t* __restrict__ xT, const bf16_t* __restrict__ Wp,
    const float* __restrict__ Wc, float* __restrict__ out) {
  const int tid  = threadIdx.x;
  const int lane = tid & 31;
  const int wave = tid >> 5;
  const int b      = blockIdx.y;
  const int o_base = (blockIdx.x * 8 + wave) * 64;   // 4 o-tiles per wave

  // 16-bit A/B VGPR layout: lanes 0-15 hold K 0-7 & 16-23, lanes 16-31 hold
  // K 8-15 & 24-31 (two aligned 16-byte chunks per lane per operand).
  const int half_sel = lane >> 4;
  const int row      = lane & 15;

  const bf16_t* a0_row = xT + ((size_t)b * C_ + row) * L_;  // c = row
  const bf16_t* a1_row = a0_row + (size_t)16 * L_;          // c = row + 16
  const bf16_t* b_rows[4];
  #pragma unroll
  for (int ot = 0; ot < 4; ++ot)
    b_rows[ot] = Wp + (size_t)(o_base + 16 * ot + row) * L_;

  v8f acc0[4], acc1[4];
  #pragma unroll
  for (int ot = 0; ot < 4; ++ot) {
    acc0[ot] = (v8f){0.f, 0.f, 0.f, 0.f, 0.f, 0.f, 0.f, 0.f};
    acc1[ot] = (v8f){0.f, 0.f, 0.f, 0.f, 0.f, 0.f, 0.f, 0.f};
  }

  union AB { v16bf v; uint4 q[2]; };

  for (int k = 0; k < L_; k += 32) {
    const int off0 = k + half_sel * 8;
    const int off1 = off0 + 16;
    AB a0, a1;
    a0.q[0] = *(const uint4*)(a0_row + off0);
    a0.q[1] = *(const uint4*)(a0_row + off1);
    a1.q[0] = *(const uint4*)(a1_row + off0);
    a1.q[1] = *(const uint4*)(a1_row + off1);
    #pragma unroll
    for (int ot = 0; ot < 4; ++ot) {
      AB bm;
      bm.q[0] = *(const uint4*)(b_rows[ot] + off0);
      bm.q[1] = *(const uint4*)(b_rows[ot] + off1);
      acc0[ot] = __builtin_amdgcn_wmma_f32_16x16x32_bf16(
          false, a0.v, false, bm.v, (short)0, acc0[ot], false, false);
      acc1[ot] = __builtin_amdgcn_wmma_f32_16x16x32_bf16(
          false, a1.v, false, bm.v, (short)0, acc1[ot], false, false);
    }
  }

  // Epilogue. C/D layout: lane holds column o = o_tile_base + (lane&15);
  // VGPR r of acc0 is row c = 8*(lane>>4) + r, acc1 adds +16.
  const int chi = half_sel * 8;
  #pragma unroll
  for (int ot = 0; ot < 4; ++ot) {
    const int o = o_base + 16 * ot + row;
    const float* wc = Wc + (size_t)o * C_;
    float s = 0.f;
    #pragma unroll
    for (int r = 0; r < 8; ++r) {
      s += acc0[ot][r] * wc[chi + r];
      s += acc1[ot][r] * wc[16 + chi + r];
    }
    s += __shfl_xor(s, 16, 32);  // fold lane pairs (l, l+16): full sum over c
    if (lane < 16) out[(size_t)b * O_ + o] = s;
  }
}

// ---------------------------------------------------------------------------
extern "C" void kernel_launch(void* const* d_in, const int* in_sizes, int n_in,
                              void* d_out, int out_size, void* d_ws, size_t ws_size,
                              hipStream_t stream) {
  const float* x      = (const float*)d_in[0];  // (B, L, C) f32
  const float* W_pos  = (const float*)d_in[1];  // (O, L)    f32
  const float* W_chan = (const float*)d_in[2];  // (O, C)    f32
  float* out = (float*)d_out;                   // (B, O)    f32

  // Workspace layout: xT bf16 (B*C*L) then W_pos bf16 (O*L)  (~143 MB total)
  bf16_t* xT  = (bf16_t*)d_ws;
  bf16_t* Wpb = (bf16_t*)((char*)d_ws + (size_t)B_ * C_ * L_ * sizeof(bf16_t));

  dim3 g1(L_ / 64, B_);
  convert_x_kernel<<<g1, 256, 0, stream>>>(x, xT);

  const int nw = O_ * L_;
  convert_w_kernel<<<(nw + 255) / 256, 256, 0, stream>>>(W_pos, Wpb, nw);

  dim3 g2(O_ / 512, B_);  // 8 waves/block, 64 o-columns per wave
  sepfc_main_kernel<<<g2, 256, 0, stream>>>(xT, Wpb, W_chan, out);
}
